// EGT_62491774156872
// MI455X (gfx1250) — matrix operand order repaired and checked by reading, not compiled, not run on hardware
//
#include <hip/hip_runtime.h>
#include <hip/hip_bf16.h>
#include <math.h>

// ---------------------------------------------------------------------------
// EGT layer forward for MI455X (gfx1250, wave32, WMMA 16x16x32 f16).
// Bandwidth-bound (~0.8 GB HBM traffic @ 23.3 TB/s ~= 35us floor).
// - Every WMMA fragment is two 16B vector loads (B-matrices pre-transposed).
// - Edge E/G projections done with WMMA via packed [WE|WG] 64x16 B-matrix.
// - Score/gate intermediates stored f16 to halve their HBM traffic.
// - Edge FFN fully fused in LDS (no 200MB of HBM intermediates).
// - Fast HW rcp/rsq instead of IEEE div/sqrt sequences in LN/sigmoid/softmax.
// ---------------------------------------------------------------------------

typedef __attribute__((ext_vector_type(16))) _Float16 v16h;
typedef __attribute__((ext_vector_type(8)))  _Float16 v8h;
typedef __attribute__((ext_vector_type(8)))  float    v8f;

#define Bz 8
#define Nz 256
#define NWz 256
#define EWz 64
#define Hz 8
#define Dz 32

__device__ __forceinline__ float fast_rcp(float x)  { return __builtin_amdgcn_rcpf(x); }
__device__ __forceinline__ float fast_rsq(float x)  { return __builtin_amdgcn_rsqf(x); }
__device__ __forceinline__ float eluf(float x)      { return x > 0.f ? x : (__expf(x) - 1.f); }
__device__ __forceinline__ float sigmoidf_(float x) { return fast_rcp(1.f + __expf(-x)); }

__device__ __forceinline__ v8f wmma_f16(v16h a, v16h b, v8f c) {
  // 8 args: (neg_a, A, neg_b, B, c_mod, C, reuse_a, reuse_b)
  return __builtin_amdgcn_wmma_f32_16x16x32_f16(false, a, false, b, (short)0, c, false, false);
}

// A fragment: 16x32 tile, row-major, stride lda (multiple of 8 halfs).
// ISA 7.12.2: lane row = lane&15; lanes<16 hold K {0..7,16..23}, lanes>=16 +8.
__device__ __forceinline__ v16h load_a(const _Float16* A, int lda, int lane) {
  int row = lane & 15, hi = lane >> 4;
  const _Float16* p = A + row * lda + (hi << 3);
  v8h lo = *(const v8h*)(p);
  v8h hh = *(const v8h*)(p + 16);
  return __builtin_shufflevector(lo, hh, 0, 1, 2, 3, 4, 5, 6, 7,
                                 8, 9, 10, 11, 12, 13, 14, 15);
}

// B fragment from TRANSPOSED storage: BT is NxK row-major (B = BT^T).
// Lane col = lane&15; lanes<16 hold K {0..7,16..23}, lanes>=16 +8.
__device__ __forceinline__ v16h load_bt(const _Float16* BT, int ldbt, int lane) {
  int col = lane & 15, hi = lane >> 4;
  const _Float16* p = BT + col * ldbt + (hi << 3);
  v8h lo = *(const v8h*)(p);
  v8h hh = *(const v8h*)(p + 16);
  return __builtin_shufflevector(lo, hh, 0, 1, 2, 3, 4, 5, 6, 7,
                                 8, 9, 10, 11, 12, 13, 14, 15);
}

// ---------------------------------------------------------------------------
// K0: convert GEMM weights to f16, transposed to N-major (one-time, ~1MB,
// permanently L2-resident). Also packs WEGt = [WE | WG]^T (16x64) so the
// edge E/G projections become a single WMMA B-matrix.
// ---------------------------------------------------------------------------
__global__ void k_prep(const float* Wqkv, const float* WOh, const float* Wh1,
                       const float* Wh2, const float* We1, const float* We2,
                       const float* WE, const float* WG, _Float16* dst) {
  int i = blockIdx.x * 256 + threadIdx.x;  // 541696 total
  float val;
  if (i < 540672) {
    const float* src;
    int li, K, Ncols;
    if (i < 196608)      { src = Wqkv; li = i;          K = 256; Ncols = 768; }
    else if (i < 262144) { src = WOh;  li = i - 196608; K = 256; Ncols = 256; }
    else if (i < 393216) { src = Wh1;  li = i - 262144; K = 256; Ncols = 512; }
    else if (i < 524288) { src = Wh2;  li = i - 393216; K = 512; Ncols = 256; }
    else if (i < 532480) { src = We1;  li = i - 524288; K = 64;  Ncols = 128; }
    else                 { src = We2;  li = i - 532480; K = 128; Ncols = 64;  }
    int n = li / K, k = li % K;
    val = src[k * Ncols + n];                       // WT[n][k] = W[k][n]
  } else {
    int li = i - 540672;                            // WEGt, 16x64
    int n = li / 64, k = li % 64;
    val = (n < 8) ? WE[k * 8 + n] : WG[k * 8 + (n - 8)];
  }
  dst[i] = (_Float16)val;
}

// ---------------------------------------------------------------------------
// Row LayerNorm over 256 cols, one wave32 per row, f16 output for WMMA A.
// ---------------------------------------------------------------------------
__global__ void k_ln256(const float* X, const float* g, const float* bta,
                        _Float16* Y, int rows) {
  int row  = blockIdx.x * 8 + (threadIdx.x >> 5);
  int lane = threadIdx.x & 31;
  if (row >= rows) return;
  const float* x = X + (long)row * 256;
  float v[8], s = 0.f, s2 = 0.f;
#pragma unroll
  for (int i = 0; i < 8; ++i) { v[i] = x[lane + 32 * i]; s += v[i]; s2 += v[i] * v[i]; }
#pragma unroll
  for (int m = 16; m >= 1; m >>= 1) { s += __shfl_xor(s, m, 32); s2 += __shfl_xor(s2, m, 32); }
  float mu = s * (1.f / 256.f);
  float var = s2 * (1.f / 256.f) - mu * mu;
  float rs = fast_rsq(var + 1e-5f);
  _Float16* y = Y + (long)row * 256;
#pragma unroll
  for (int i = 0; i < 8; ++i) {
    int c = lane + 32 * i;
    y[c] = (_Float16)((v[i] - mu) * rs * g[c] + bta[c]);
  }
}

// ---------------------------------------------------------------------------
// K3: edge E/G projection via WMMA. Block = 128 edge rows:
//   stage1: LN(e) rows -> LDS f16 (2 threads/row, shfl-width-2 reduction)
//   stage2: one 16-row tile per wave, B = WEGt (cols 0-7 = WE, 8-15 = WG)
//           epilogue: E+bE -> Ebuf f16 ; sigmoid(G+bG) -> Gate f16
// ---------------------------------------------------------------------------
__global__ void k_eproj(const float* e, const float* g, const float* bln,
                        const _Float16* WEGt, const float* bE, const float* bG,
                        _Float16* Ebuf, _Float16* Gate) {
  __shared__ alignas(16) _Float16 As[128 * 64];   // 16 KB LN rows
  __shared__ float sg[64], sb[64];
  int t = threadIdx.x;
  if (t < 64) { sg[t] = g[t]; sb[t] = bln[t]; }
  __syncthreads();

  long rows0 = (long)blockIdx.x * 128;
  int rl = t >> 1, part = t & 1;          // 2 threads per edge row
  long row = rows0 + rl;
  const float* er = e + row * 64 + part * 32;
  float x[32], s = 0.f, s2 = 0.f;
#pragma unroll
  for (int j = 0; j < 32; ++j) { x[j] = er[j]; s += x[j]; s2 += x[j] * x[j]; }
  s  += __shfl_xor(s, 1, 2);
  s2 += __shfl_xor(s2, 1, 2);
  float mu = s * (1.f / 64.f);
  float var = s2 * (1.f / 64.f) - mu * mu;
  float rs = fast_rsq(var + 1e-5f);
#pragma unroll
  for (int j = 0; j < 32; ++j) {
    int c = part * 32 + j;
    As[rl * 64 + c] = (_Float16)((x[j] - mu) * rs * sg[c] + sb[c]);
  }
  __syncthreads();

  int lane = t & 31, w = t >> 5;          // wave w handles rows [w*16, w*16+16)
  v8f acc = {};
  for (int k0 = 0; k0 < 64; k0 += 32)
    acc = wmma_f16(load_a(As + w * 16 * 64 + k0, 64, lane),
                   load_bt(WEGt + k0, 64, lane), acc);
  int col = lane & 15, hi = lane >> 4;
  int h = col & 7;
  float bias = (col < 8) ? bE[h] : bG[h];
#pragma unroll
  for (int j = 0; j < 8; ++j) {
    long r = rows0 + w * 16 + j + (hi << 3);   // global edge row
    long b = r >> 16, lm = r & 65535;
    long o = ((b * 8 + h) << 16) + lm;
    float v = acc[j] + bias;
    if (col < 8) Ebuf[o] = (_Float16)v;
    else         Gate[o] = (_Float16)sigmoidf_(v);
  }
}

// ---------------------------------------------------------------------------
// K4: qkv = h_ln @ Wqkv + bqkv; Q,K scattered to (B,H,N,D) f16,
// V scattered TRANSPOSED to (B,H,D,N) f16 so the V-att GEMM reads it as BT.
// ---------------------------------------------------------------------------
__global__ void k_qkv(const _Float16* Aln, const _Float16* WT, const float* bias,
                      _Float16* Qh, _Float16* Kh, _Float16* VhT) {
  int lane = threadIdx.x & 31;
  int tile = blockIdx.x * 8 + (threadIdx.x >> 5);  // < 6144
  int tm = tile / 48, tn = tile % 48;
  const _Float16* Ap = Aln + (long)tm * 16 * 256;
  const _Float16* Bp = WT + (long)tn * 16 * 256;
  v8f acc = {};
  for (int k0 = 0; k0 < 256; k0 += 32)
    acc = wmma_f16(load_a(Ap + k0, 256, lane), load_bt(Bp + k0, 256, lane), acc);
  int col = lane & 15, hi = lane >> 4;
  int cg = tn * 16 + col;
  int hh = cg & 7, d = cg >> 3;  // qkv col = d*H + h, d in [0,96)
#pragma unroll
  for (int j = 0; j < 8; ++j) {
    int rg = tm * 16 + j + (hi << 3);
    int b = rg >> 8, l = rg & 255;
    _Float16 hv = (_Float16)(acc[j] + bias[cg]);
    if (d < 32)      Qh [(((long)(b * 8 + hh)) * 256 + l) * 32 + d] = hv;
    else if (d < 64) Kh [(((long)(b * 8 + hh)) * 256 + l) * 32 + (d - 32)] = hv;
    else             VhT[(((long)(b * 8 + hh)) * 32 + (d - 64)) * 256 + l] = hv;
  }
}

// ---------------------------------------------------------------------------
// K5: H_hat = clip(Q@K^T / sqrt(D), +-5) + E   (per (b,h), 256x256x32), f16
// ---------------------------------------------------------------------------
__global__ void k_scores(const _Float16* Qh, const _Float16* Kh,
                         const _Float16* Ebuf, _Float16* Hhat) {
  int lane = threadIdx.x & 31;
  int tile = blockIdx.x * 8 + (threadIdx.x >> 5);  // < 16384
  int bh = tile >> 8, tm = (tile >> 4) & 15, tn = tile & 15;
  const _Float16* Qp = Qh + (long)bh * 8192 + tm * 16 * 32;
  const _Float16* Kp = Kh + (long)bh * 8192 + tn * 16 * 32;
  v8f acc = {};
  acc = wmma_f16(load_a(Qp, 32, lane), load_bt(Kp, 32, lane), acc);  // K = D = 32
  int col = lane & 15, hi = lane >> 4;
#pragma unroll
  for (int j = 0; j < 8; ++j) {
    int l = tm * 16 + j + (hi << 3), m = tn * 16 + col;
    float v = acc[j] * 0.17677669529663687f;  // D^-0.5
    v = fminf(fmaxf(v, -5.f), 5.f);
    long o = ((long)bh << 16) + l * 256 + m;
    Hhat[o] = (_Float16)(v + (float)Ebuf[o]);
  }
}

// ---------------------------------------------------------------------------
// K6: A_tild = softmax_m(H_hat) * gate, f16 out. One wave per (b,h,l) row.
// ---------------------------------------------------------------------------
__global__ void k_softmax(const _Float16* Hhat, const _Float16* Gate, _Float16* At) {
  int row  = blockIdx.x * 8 + (threadIdx.x >> 5);  // < 16384
  int lane = threadIdx.x & 31;
  const _Float16* x = Hhat + (long)row * 256;
  float v[8], mx = -1e30f;
#pragma unroll
  for (int i = 0; i < 8; ++i) { v[i] = (float)x[lane + 32 * i]; mx = fmaxf(mx, v[i]); }
#pragma unroll
  for (int m = 16; m >= 1; m >>= 1) mx = fmaxf(mx, __shfl_xor(mx, m, 32));
  float s = 0.f;
#pragma unroll
  for (int i = 0; i < 8; ++i) { v[i] = __expf(v[i] - mx); s += v[i]; }
#pragma unroll
  for (int m = 16; m >= 1; m >>= 1) s += __shfl_xor(s, m, 32);
  float inv = fast_rcp(s);
  const _Float16* gp = Gate + (long)row * 256;
  _Float16* yp = At + (long)row * 256;
#pragma unroll
  for (int i = 0; i < 8; ++i) {
    int c = lane + 32 * i;
    yp[c] = (_Float16)(v[i] * inv * (float)gp[c]);
  }
}

// ---------------------------------------------------------------------------
// K7: V_att = A_tild @ V per (b,h) (256x32x256); V read from (B,H,D,N) as BT.
// Result scattered to (B,N,256) f16 with col = d*8 + h (reference reshape).
// ---------------------------------------------------------------------------
__global__ void k_vatt(const _Float16* At, const _Float16* VhT, _Float16* Vatt) {
  int lane = threadIdx.x & 31;
  int tile = blockIdx.x * 8 + (threadIdx.x >> 5);  // < 2048
  int bh = tile >> 5, tm = (tile >> 1) & 15, tn = tile & 1;
  const _Float16* Ap = At + ((long)bh << 16) + tm * 16 * 256;
  const _Float16* Bp = VhT + (long)bh * 8192 + (long)tn * 16 * 256;  // (D,N)
  v8f acc = {};
  for (int k0 = 0; k0 < 256; k0 += 32)
    acc = wmma_f16(load_a(Ap + k0, 256, lane), load_bt(Bp + k0, 256, lane), acc);
  int col = lane & 15, hi = lane >> 4;
  int b = bh >> 3, hh = bh & 7, d = tn * 16 + col;
#pragma unroll
  for (int j = 0; j < 8; ++j) {
    int l = tm * 16 + j + (hi << 3);
    Vatt[((long)b * 256 + l) * 256 + d * 8 + hh] = (_Float16)acc[j];
  }
}

// ---------------------------------------------------------------------------
// K8: h1 = V_att @ WOh + bOh + h   (2048x256x256), WOh read transposed
// ---------------------------------------------------------------------------
__global__ void k_oh(const _Float16* Vatt, const _Float16* WT, const float* bias,
                     const float* hin, float* h1) {
  int lane = threadIdx.x & 31;
  int tile = blockIdx.x * 8 + (threadIdx.x >> 5);  // < 2048
  int tm = tile >> 4, tn = tile & 15;
  const _Float16* Ap = Vatt + (long)tm * 16 * 256;
  const _Float16* Bp = WT + (long)tn * 16 * 256;
  v8f acc = {};
  for (int k0 = 0; k0 < 256; k0 += 32)
    acc = wmma_f16(load_a(Ap + k0, 256, lane), load_bt(Bp + k0, 256, lane), acc);
  int col = lane & 15, hi = lane >> 4, cg = tn * 16 + col;
  float bb = bias[cg];
#pragma unroll
  for (int j = 0; j < 8; ++j) {
    int rg = tm * 16 + j + (hi << 3);
    long o = (long)rg * 256 + cg;
    h1[o] = acc[j] + bb + hin[o];
  }
}

// ---------------------------------------------------------------------------
// K10: ff1 = elu(h1_ln @ Wh1 + bh1)   (2048x512x256), Wh1 transposed, f16 out
// ---------------------------------------------------------------------------
__global__ void k_ff1(const _Float16* A, const _Float16* WT, const float* bias,
                      _Float16* Y) {
  int lane = threadIdx.x & 31;
  int tile = blockIdx.x * 8 + (threadIdx.x >> 5);  // < 4096
  int tm = tile >> 5, tn = tile & 31;
  const _Float16* Ap = A + (long)tm * 16 * 256;
  const _Float16* Bp = WT + (long)tn * 16 * 256;
  v8f acc = {};
  for (int k0 = 0; k0 < 256; k0 += 32)
    acc = wmma_f16(load_a(Ap + k0, 256, lane), load_bt(Bp + k0, 256, lane), acc);
  int col = lane & 15, hi = lane >> 4, cg = tn * 16 + col;
  float bb = bias[cg];
#pragma unroll
  for (int j = 0; j < 8; ++j) {
    int rg = tm * 16 + j + (hi << 3);
    Y[(long)rg * 512 + cg] = (_Float16)eluf(acc[j] + bb);
  }
}

// ---------------------------------------------------------------------------
// K11: h_out = ff1 @ Wh2 + bh2 + h1   (2048x256x512), Wh2 transposed
// ---------------------------------------------------------------------------
__global__ void k_ff2(const _Float16* A, const _Float16* WT, const float* bias,
                      const float* res, float* Y) {
  int lane = threadIdx.x & 31;
  int tile = blockIdx.x * 8 + (threadIdx.x >> 5);  // < 2048
  int tm = tile >> 4, tn = tile & 15;
  const _Float16* Ap = A + (long)tm * 16 * 512;
  const _Float16* Bp = WT + (long)tn * 16 * 512;
  v8f acc = {};
  for (int k0 = 0; k0 < 512; k0 += 32)
    acc = wmma_f16(load_a(Ap + k0, 512, lane), load_bt(Bp + k0, 512, lane), acc);
  int col = lane & 15, hi = lane >> 4, cg = tn * 16 + col;
  float bb = bias[cg];
#pragma unroll
  for (int j = 0; j < 8; ++j) {
    int rg = tm * 16 + j + (hi << 3);
    long o = (long)rg * 256 + cg;
    Y[o] = acc[j] + bb + res[o];
  }
}

// ---------------------------------------------------------------------------
// K12 (fused edge path): per block of 64 edge rows:
//   eo  = e + bOe + Hrow@WOe      (residual, LDS f32)
//   ln  = LN(eo)                  (f16 WMMA A in LDS)
//   act = elu(ln @ We1 + be1)     (WMMA, f16 in LDS)
//   e_out = act @ We2 + be2 + eo  (WMMA)
// ---------------------------------------------------------------------------
__global__ void k_effn(const float* e, const _Float16* Hhat, const float* WOe,
                       const float* bOe, const float* g2, const float* b2,
                       const _Float16* We1T, const float* be1,
                       const _Float16* We2T, const float* be2, float* eout) {
  __shared__ float sWOe[512], sbOe[64], sg[64], sb[64];
  __shared__ float eo_s[64 * 64];                     // 16 KB residual
  __shared__ alignas(16) _Float16 As[64 * 64];        // 8 KB  LN rows
  __shared__ alignas(16) _Float16 acts[64 * 128];     // 16 KB hidden acts
  int t = threadIdx.x;
  for (int i = t; i < 512; i += 256) sWOe[i] = WOe[i];
  if (t < 64) { sbOe[t] = bOe[t]; sg[t] = g2[t]; sb[t] = b2[t]; }
  __syncthreads();

  long rows0 = (long)blockIdx.x * 64;
  int rl = t >> 2, part = t & 3;        // 4 threads per edge row
  long row = rows0 + rl;
  long b = row >> 16, lm = row & 65535;

  float hrow[8];
#pragma unroll
  for (int h = 0; h < 8; ++h) hrow[h] = (float)Hhat[((b * 8 + h) << 16) + lm];

  float eo[16], s = 0.f, s2 = 0.f;
#pragma unroll
  for (int j = 0; j < 16; ++j) {
    int c = part * 16 + j;
    float v = e[row * 64 + c] + sbOe[c];
#pragma unroll
    for (int h = 0; h < 8; ++h) v += hrow[h] * sWOe[h * 64 + c];
    eo[j] = v; eo_s[rl * 64 + c] = v; s += v; s2 += v * v;
  }
  s  += __shfl_xor(s, 1, 4);  s  += __shfl_xor(s, 2, 4);
  s2 += __shfl_xor(s2, 1, 4); s2 += __shfl_xor(s2, 2, 4);
  float mu = s * (1.f / 64.f);
  float var = s2 * (1.f / 64.f) - mu * mu;
  float rs = fast_rsq(var + 1e-5f);
#pragma unroll
  for (int j = 0; j < 16; ++j) {
    int c = part * 16 + j;
    As[rl * 64 + c] = (_Float16)((eo[j] - mu) * rs * sg[c] + sb[c]);
  }
  __syncthreads();

  int lane = t & 31, w = t >> 5;
  // GEMM1: (64x64) @ We1(64x128) -> elu -> acts ; 32 tiles over 8 waves
#pragma unroll
  for (int tt = 0; tt < 4; ++tt) {
    int tile = w + tt * 8;
    int tm = tile >> 3, tn = tile & 7;
    v8f acc = {};
    for (int k0 = 0; k0 < 64; k0 += 32)
      acc = wmma_f16(load_a(As + tm * 16 * 64 + k0, 64, lane),
                     load_bt(We1T + tn * 16 * 64 + k0, 64, lane), acc);
    int col = lane & 15, hi = lane >> 4, cg = tn * 16 + col;
    float bb = be1[cg];
#pragma unroll
    for (int j = 0; j < 8; ++j) {
      int r = tm * 16 + j + (hi << 3);
      acts[r * 128 + cg] = (_Float16)eluf(acc[j] + bb);
    }
  }
  __syncthreads();
  // GEMM2: (64x128) @ We2(128x64) + be2 + eo -> e_out ; 16 tiles over 8 waves
#pragma unroll
  for (int tt = 0; tt < 2; ++tt) {
    int tile = w + tt * 8;
    int tm = tile >> 2, tn = tile & 3;
    v8f acc = {};
    for (int k0 = 0; k0 < 128; k0 += 32)
      acc = wmma_f16(load_a(acts + tm * 16 * 128 + k0, 128, lane),
                     load_bt(We2T + tn * 16 * 128 + k0, 128, lane), acc);
    int col = lane & 15, hi = lane >> 4, cg = tn * 16 + col;
    float bb = be2[cg];
#pragma unroll
    for (int j = 0; j < 8; ++j) {
      int r = tm * 16 + j + (hi << 3);
      eout[(rows0 + r) * 64 + cg] = acc[j] + bb + eo_s[r * 64 + cg];
    }
  }
}

// ---------------------------------------------------------------------------
extern "C" void kernel_launch(void* const* d_in, const int* in_sizes, int n_in,
                              void* d_out, int out_size, void* d_ws, size_t ws_size,
                              hipStream_t stream) {
  const float* h     = (const float*)d_in[0];
  const float* e     = (const float*)d_in[1];
  const float* ln1hg = (const float*)d_in[2];
  const float* ln1hb = (const float*)d_in[3];
  const float* ln1eg = (const float*)d_in[4];
  const float* ln1eb = (const float*)d_in[5];
  const float* Wqkv  = (const float*)d_in[6];
  const float* bqkv  = (const float*)d_in[7];
  const float* WE    = (const float*)d_in[8];
  const float* bE    = (const float*)d_in[9];
  const float* WG    = (const float*)d_in[10];
  const float* bG    = (const float*)d_in[11];
  const float* WOh   = (const float*)d_in[12];
  const float* bOh   = (const float*)d_in[13];
  const float* ln2hg = (const float*)d_in[14];
  const float* ln2hb = (const float*)d_in[15];
  const float* Wh1   = (const float*)d_in[16];
  const float* bh1   = (const float*)d_in[17];
  const float* Wh2   = (const float*)d_in[18];
  const float* bh2   = (const float*)d_in[19];
  const float* WOe   = (const float*)d_in[20];
  const float* bOe   = (const float*)d_in[21];
  const float* ln2eg = (const float*)d_in[22];
  const float* ln2eb = (const float*)d_in[23];
  const float* We1   = (const float*)d_in[24];
  const float* be1   = (const float*)d_in[25];
  const float* We2   = (const float*)d_in[26];
  const float* be2   = (const float*)d_in[27];

  float* h_out = (float*)d_out;                 // (8,256,256)
  float* e_out = h_out + (long)Bz * Nz * NWz;   // (8,256,256,64)

  // workspace bump allocator (256B aligned), ~45 MB total
  char* p = (char*)d_ws;
  auto alloc = [&](size_t bytes) -> void* {
    void* r = (void*)p;
    p += (bytes + 255) & ~(size_t)255;
    return r;
  };
  _Float16* Wf   = (_Float16*)alloc(541696 * 2);     // f16 transposed weights
  _Float16* hln  = (_Float16*)alloc(524288 * 2);     // LN(h) f16
  _Float16* Qh   = (_Float16*)alloc(524288 * 2);     // (B,H,N,D)
  _Float16* Kh   = (_Float16*)alloc(524288 * 2);     // (B,H,N,D)
  _Float16* VhT  = (_Float16*)alloc(524288 * 2);     // (B,H,D,N)
  _Float16* Ebuf = (_Float16*)alloc(4194304ull * 2); // (B,H,N,N) f16
  _Float16* Gate = (_Float16*)alloc(4194304ull * 2);
  _Float16* Hhat = (_Float16*)alloc(4194304ull * 2);
  _Float16* At   = (_Float16*)alloc(4194304ull * 2);
  _Float16* Vatt = (_Float16*)alloc(524288 * 2);     // (B,N,256) col=d*8+h
  float*    h1   = (float*)alloc(524288 * 4);
  _Float16* h1ln = (_Float16*)alloc(524288 * 2);
  _Float16* ff1  = (_Float16*)alloc(1048576ull * 2);

  _Float16* WqkvT = Wf;
  _Float16* WOhT  = Wf + 196608;
  _Float16* Wh1T  = Wf + 262144;
  _Float16* Wh2T  = Wf + 393216;
  _Float16* We1T  = Wf + 524288;
  _Float16* We2T  = Wf + 532480;
  _Float16* WEGt  = Wf + 540672;

  k_prep   <<<2116, 256, 0, stream>>>(Wqkv, WOh, Wh1, Wh2, We1, We2, WE, WG, Wf);
  k_ln256  <<<256,  256, 0, stream>>>(h, ln1hg, ln1hb, hln, 2048);
  k_eproj  <<<4096, 256, 0, stream>>>(e, ln1eg, ln1eb, WEGt, bE, bG, Ebuf, Gate);
  k_qkv    <<<768,  256, 0, stream>>>(hln, WqkvT, bqkv, Qh, Kh, VhT);
  k_scores <<<2048, 256, 0, stream>>>(Qh, Kh, Ebuf, Hhat);
  k_softmax<<<2048, 256, 0, stream>>>(Hhat, Gate, At);
  k_vatt   <<<256,  256, 0, stream>>>(At, VhT, Vatt);
  k_oh     <<<256,  256, 0, stream>>>(Vatt, WOhT, bOh, h, h1);
  k_ln256  <<<256,  256, 0, stream>>>(h1, ln2hg, ln2hb, h1ln, 2048);
  k_ff1    <<<512,  256, 0, stream>>>(h1ln, Wh1T, bh1, ff1);
  k_ff2    <<<256,  256, 0, stream>>>(ff1, Wh2T, bh2, h1, h_out);
  k_effn   <<<8192, 256, 0, stream>>>(e, Hhat, WOe, bOe, ln2eg, ln2eb,
                                      We1T, be1, We2T, be2, e_out);
}
